// Attention_30107720745648
// MI455X (gfx1250) — matrix operand order repaired and checked by reading, not compiled, not run on hardware
//
#include <hip/hip_runtime.h>
#include <hip/hip_bf16.h>

// ---------------------------------------------------------------------------
// Problem constants (from reference)
// ---------------------------------------------------------------------------
#define BATCH    4
#define SEQ      2048
#define DIM      1024
#define HEADS    16
#define DHEAD    64
#define NNULL    2
#define JTOT     2050          // SEQ + NNULL
#define JPAD     2080          // padded key stride for v_t (32B-aligned rows)
#define MTOT     (BATCH*SEQ)   // 8192 token rows
#define NEG_BIG  (-3.0e38f)

typedef __bf16 v16bf __attribute__((ext_vector_type(16)));
typedef float  v8f   __attribute__((ext_vector_type(8)));
typedef unsigned int u32x4 __attribute__((ext_vector_type(4)));

union Frag16 { v16bf bf; u32x4 q[2]; };

__device__ __forceinline__ v16bf load_pair(const __bf16* p0, const __bf16* p1) {
    Frag16 f;
    f.q[0] = *(const u32x4*)p0;   // 16 bytes = 8 bf16
    f.q[1] = *(const u32x4*)p1;
    return f.bf;
}

__device__ __forceinline__ unsigned short f2bf(float x) {
    __bf16 h = (__bf16)x;
    return __builtin_bit_cast(unsigned short, h);
}

__device__ __forceinline__ v8f wmma_bf16(v16bf a, v16bf b, v8f c) {
    return __builtin_amdgcn_wmma_f32_16x16x32_bf16(false, a, false, b, (short)0, c, false, false);
}

// ---------------------------------------------------------------------------
// DPP16 butterfly reductions within each 16-lane row (pure VALU, no LDS).
//   0xB1 = quad_perm(1,0,3,2)  (xor 1)
//   0x4E = quad_perm(2,3,0,1)  (xor 2)
//   0x141 = row_half_mirror    (combines quads within 8)
//   0x140 = row_mirror         (combines 8-groups within 16)
// After all 4 steps every lane of the 16-lane row holds the full reduction.
// ---------------------------------------------------------------------------
template <int CTRL>
__device__ __forceinline__ float dpp_movf(float x) {
    int r = __builtin_amdgcn_update_dpp(0, __builtin_bit_cast(int, x),
                                        CTRL, 0xF, 0xF, true);
    return __builtin_bit_cast(float, r);
}

__device__ __forceinline__ float red_max16(float x) {
    x = fmaxf(x, dpp_movf<0xB1>(x));
    x = fmaxf(x, dpp_movf<0x4E>(x));
    x = fmaxf(x, dpp_movf<0x141>(x));
    x = fmaxf(x, dpp_movf<0x140>(x));
    return x;
}

__device__ __forceinline__ float red_sum16(float x) {
    x += dpp_movf<0xB1>(x);
    x += dpp_movf<0x4E>(x);
    x += dpp_movf<0x141>(x);
    x += dpp_movf<0x140>(x);
    return x;
}

// ---------------------------------------------------------------------------
// 1) fp32 -> bf16 transposed weight conversion: dst[n*K + k] = src[k*N + n]
// ---------------------------------------------------------------------------
__global__ void convT_kernel(const float* __restrict__ src, __bf16* __restrict__ dst,
                             int K, int N) {
    long idx = (long)blockIdx.x * 256 + threadIdx.x;
    long total = (long)K * N;
    if (idx >= total) return;
    int k = (int)(idx / N);
    int n = (int)(idx % N);
    dst[(long)n * K + k] = (__bf16)src[idx];
}

// ---------------------------------------------------------------------------
// 2) LayerNorm over DIM, output bf16
// ---------------------------------------------------------------------------
__global__ void ln_kernel(const float* __restrict__ x, const float* __restrict__ gamma,
                          __bf16* __restrict__ xn) {
    const int row = blockIdx.x;
    const int tid = threadIdx.x;
    const float* xr = x + (size_t)row * DIM;
    float s = 0.f, ss = 0.f;
    for (int c = tid; c < DIM; c += 256) { float v = xr[c]; s += v; ss += v * v; }
    __shared__ float rs[256], rss[256];
    rs[tid] = s; rss[tid] = ss;
    __syncthreads();
    for (int o = 128; o > 0; o >>= 1) {
        if (tid < o) { rs[tid] += rs[tid + o]; rss[tid] += rss[tid + o]; }
        __syncthreads();
    }
    const float mean = rs[0] * (1.0f / DIM);
    const float var  = rss[0] * (1.0f / DIM) - mean * mean;
    const float inv  = rsqrtf(var + 1e-5f);
    for (int c = tid; c < DIM; c += 256)
        xn[(size_t)row * DIM + c] = (__bf16)((xr[c] - mean) * inv * gamma[c]);
}

// ---------------------------------------------------------------------------
// 3) WMMA GEMM: C[M,N] = A[M,K] (bf16, row-major) x BT[N,K] (bf16 transposed)
//    One wave computes a 16(M) x 64(N) tile. MODE selects the fused epilogue:
//      0: plain f32 store to Cout
//      1: q: per-row l2norm * q_scale -> bf16 q_n[b,h,i,d]
//      2: kv: cols<1024 -> k (l2norm*k_scale) into k_n[b,h,2+i,d];
//             cols>=1024 -> v bf16 into v_t[b,h,d,2+i] (transposed, JPAD stride)
// ---------------------------------------------------------------------------
template <int MODE>
__global__ void gemm_wmma_kernel(const __bf16* __restrict__ A, const __bf16* __restrict__ BT,
                                 int K, int Ncols, float* __restrict__ Cout,
                                 __bf16* __restrict__ q_n, __bf16* __restrict__ k_n,
                                 __bf16* __restrict__ v_t,
                                 const float* __restrict__ qscale,
                                 const float* __restrict__ kscale) {
    const int lane = threadIdx.x & 31;
    const int half = lane >> 4;
    const int c    = lane & 15;
    const int wid  = blockIdx.x * (blockDim.x >> 5) + (threadIdx.x >> 5);
    const int ntiles = Ncols >> 6;
    const int mtiles = MTOT >> 4;
    if (wid >= mtiles * ntiles) return;          // wave-uniform exit
    const int mt = wid / ntiles, nb = wid % ntiles;
    const int m0 = mt << 4, n0 = nb << 6;

    const __bf16* arow = A + (size_t)(m0 + c) * K;
    const __bf16* brow[4];
#pragma unroll
    for (int a = 0; a < 4; a++) brow[a] = BT + (size_t)(n0 + a * 16 + c) * K;

    v8f acc[4];
    const v8f vzero = {0.f, 0.f, 0.f, 0.f, 0.f, 0.f, 0.f, 0.f};
#pragma unroll
    for (int a = 0; a < 4; a++) acc[a] = vzero;

    for (int k0 = 0; k0 < K; k0 += 32) {
        v16bf af = load_pair(arow + k0 + half * 8, arow + k0 + 16 + half * 8);
#pragma unroll
        for (int a = 0; a < 4; a++) {
            v16bf bfr = load_pair(brow[a] + k0 + half * 8, brow[a] + k0 + 16 + half * 8);
            acc[a] = wmma_bf16(af, bfr, acc[a]);
        }
    }

    if (MODE == 0) {
#pragma unroll
        for (int r = 0; r < 8; r++) {
            const int mg = m0 + r + half * 8;
#pragma unroll
            for (int a = 0; a < 4; a++)
                Cout[(size_t)mg * Ncols + n0 + a * 16 + c] = acc[a][r];
        }
    } else {
        // per-row sum of squares across the 64 columns this wave owns
        float ssq[8];
#pragma unroll
        for (int r = 0; r < 8; r++) {
            float s = 0.f;
#pragma unroll
            for (int a = 0; a < 4; a++) s += acc[a][r] * acc[a][r];
            ssq[r] = red_sum16(s);
        }
        if (MODE == 1) {
            const int head = nb;                  // Ncols==1024 -> nb in [0,16)
#pragma unroll
            for (int r = 0; r < 8; r++) {
                const int mg = m0 + r + half * 8;
                const int bidx = mg >> 11, i = mg & (SEQ - 1);
                const float inv = 1.0f / fmaxf(sqrtf(ssq[r]), 1e-12f);
#pragma unroll
                for (int a = 0; a < 4; a++) {
                    const int d = a * 16 + c;
                    q_n[(((size_t)bidx * HEADS + head) * SEQ + i) * DHEAD + d] =
                        (__bf16)(acc[a][r] * inv * qscale[d]);
                }
            }
        } else {  // MODE == 2 : kv
            if (nb < 16) {                        // K half
                const int head = nb;
#pragma unroll
                for (int r = 0; r < 8; r++) {
                    const int mg = m0 + r + half * 8;
                    const int bidx = mg >> 11, i = mg & (SEQ - 1);
                    const float inv = 1.0f / fmaxf(sqrtf(ssq[r]), 1e-12f);
#pragma unroll
                    for (int a = 0; a < 4; a++) {
                        const int d = a * 16 + c;
                        k_n[(((size_t)bidx * HEADS + head) * JTOT + NNULL + i) * DHEAD + d] =
                            (__bf16)(acc[a][r] * inv * kscale[d]);
                    }
                }
            } else {                              // V half, stored transposed
                const int head = nb - 16;
#pragma unroll
                for (int r = 0; r < 8; r++) {
                    const int mg = m0 + r + half * 8;
                    const int bidx = mg >> 11, i = mg & (SEQ - 1);
#pragma unroll
                    for (int a = 0; a < 4; a++) {
                        const int d = a * 16 + c;
                        v_t[(((size_t)bidx * HEADS + head) * DHEAD + d) * JPAD + NNULL + i] =
                            (__bf16)acc[a][r];
                    }
                }
            }
        }
    }
}

// ---------------------------------------------------------------------------
// 4) Null-KV fill: normalized k nulls at j=0,1 ; raw v nulls (transposed).
//    One wave per (b,h,nn); each 16-lane half redundantly computes the full
//    64-dim norm (4 elems/lane) so the reduction stays within a DPP16 row.
// ---------------------------------------------------------------------------
__global__ void nullkv_kernel(const float* __restrict__ null_kv,
                              const float* __restrict__ k_scale,
                              __bf16* __restrict__ k_n, __bf16* __restrict__ v_t) {
    const int gid = blockIdx.x * 256 + threadIdx.x;
    const int row = gid >> 5, lane = gid & 31;
    const int half = lane >> 4, c = lane & 15;
    if (row >= BATCH * HEADS * NNULL) return;
    const int b = row >> 5, h = (row >> 1) & 15, nn = row & 1;
    // null_kv layout (H, NNULL, 2, D): [:,:,0]=k, [:,:,1]=v
    const float* nk = null_kv + (((size_t)h * NNULL + nn) * 2 + 0) * DHEAD;
    const float* nv = null_kv + (((size_t)h * NNULL + nn) * 2 + 1) * DHEAD;
    float s = 0.f;
#pragma unroll
    for (int e = 0; e < 4; e++) { float v = nk[c * 4 + e]; s += v * v; }
    s = red_sum16(s);
    const float inv = 1.0f / fmaxf(sqrtf(s), 1e-12f);
    if (half == 0) {
        const size_t kb = (((size_t)b * HEADS + h) * JTOT + nn) * DHEAD;
        const size_t vb = ((size_t)b * HEADS + h) * DHEAD * (size_t)JPAD;
#pragma unroll
        for (int e = 0; e < 4; e++) {
            const int d = c * 4 + e;
            k_n[kb + d] = (__bf16)(nk[d] * inv * k_scale[d]);
            v_t[vb + (size_t)d * JPAD + nn] = (__bf16)nv[d];
        }
    }
}

// ---------------------------------------------------------------------------
// 5) Flash attention: one wave per 16 query rows of one (b,h).
//    S = QK^T via 2 chained WMMAs; scale/ALiBi/causal in-register; online
//    softmax (DPP reductions); P via LDS -> A-frag; PV via 4 WMMAs.
// ---------------------------------------------------------------------------
__global__ void attn_kernel(const __bf16* __restrict__ q_n, const __bf16* __restrict__ k_n,
                            const __bf16* __restrict__ v_t, __bf16* __restrict__ attn_out) {
    __shared__ __align__(16) unsigned short lds_p[8][16 * 32];

    const int lane = threadIdx.x & 31;
    const int half = lane >> 4;
    const int c    = lane & 15;
    const int wloc = threadIdx.x >> 5;
    const int w    = blockIdx.x * 8 + wloc;
    const int it = w & 127, h = (w >> 7) & 15, b = w >> 11;
    const int i0 = it << 4;
    const float slope = exp2f(-0.5f * (float)(h + 1));   // 16-head ALiBi slopes

    // Q A-fragments (K = d : two 32-wide steps)
    const __bf16* qrow = q_n + (((size_t)b * HEADS + h) * SEQ + i0 + c) * DHEAD;
    const v16bf qa0 = load_pair(qrow + half * 8,      qrow + 16 + half * 8);
    const v16bf qa1 = load_pair(qrow + 32 + half * 8, qrow + 48 + half * 8);

    const __bf16* kbase = k_n + ((size_t)b * HEADS + h) * (size_t)JTOT * DHEAD;
    const __bf16* vbase = v_t + ((size_t)b * HEADS + h) * (size_t)DHEAD * JPAD;

    const v8f vzero = {0.f, 0.f, 0.f, 0.f, 0.f, 0.f, 0.f, 0.f};
    v8f acc[4];
#pragma unroll
    for (int a = 0; a < 4; a++) acc[a] = vzero;
    float m_run[8], l_run[8];
#pragma unroll
    for (int r = 0; r < 8; r++) { m_run[r] = NEG_BIG; l_run[r] = 0.f; }

    unsigned short* myl = lds_p[wloc];
    const int ntile = (i0 + 15 + NNULL) / 32 + 1;        // causal: skip masked tiles

    for (int t = 0; t < ntile; t++) {
        const int jt = t * 32;
        // ---- S tile (16 x 32) = Q (16x64) . K^T ----
        v8f s[2];
#pragma unroll
        for (int a = 0; a < 2; a++) {
            int j = jt + a * 16 + c;
            int jc = j < JTOT ? j : (JTOT - 1);           // clamp (value masked anyway)
            const __bf16* krow = kbase + (size_t)jc * DHEAD;
            v16bf kb0 = load_pair(krow + half * 8,      krow + 16 + half * 8);
            v16bf kb1 = load_pair(krow + 32 + half * 8, krow + 48 + half * 8);
            s[a] = wmma_bf16(qa0, kb0, vzero);
            s[a] = wmma_bf16(qa1, kb1, s[a]);
        }
        // ---- scale + ALiBi + causal mask + online softmax ----
#pragma unroll
        for (int r = 0; r < 8; r++) {
            const int i = i0 + r + half * 8;              // query index
            const int j0c = jt + c, j1c = jt + 16 + c;    // key indices
            float v0 = s[0][r] * 8.0f - slope * fabsf((float)(j0c - (i + NNULL)));
            float v1 = s[1][r] * 8.0f - slope * fabsf((float)(j1c - (i + NNULL)));
            if (j0c > i + NNULL || j0c >= JTOT) v0 = NEG_BIG;
            if (j1c > i + NNULL || j1c >= JTOT) v1 = NEG_BIG;
            const float mx = red_max16(fmaxf(v0, v1));
            const float mnew = fmaxf(m_run[r], mx);
            const float corr = __expf(m_run[r] - mnew);
            const float p0 = __expf(v0 - mnew);
            const float p1 = __expf(v1 - mnew);
            const float sum = red_sum16(p0 + p1);
            l_run[r] = l_run[r] * corr + sum;
            m_run[r] = mnew;
#pragma unroll
            for (int a = 0; a < 4; a++) acc[a][r] = acc[a][r] * corr;
            myl[(r + half * 8) * 32 + c]      = f2bf(p0);
            myl[(r + half * 8) * 32 + 16 + c] = f2bf(p1);
        }
        asm volatile("s_wait_dscnt 0x0" ::: "memory");
        __builtin_amdgcn_wave_barrier();
        // ---- P (A-frag from LDS) x V (B-frags from transposed v_t) ----
        const __bf16* prow = (const __bf16*)myl + c * 32;
        v16bf pa = load_pair(prow + half * 16, prow + half * 16 + 8);
#pragma unroll
        for (int a = 0; a < 4; a++) {
            const __bf16* vrow = vbase + (size_t)(a * 16 + c) * JPAD + jt + half * 16;
            v16bf vb = load_pair(vrow, vrow + 8);
            acc[a] = wmma_bf16(pa, vb, acc[a]);
        }
        asm volatile("s_wait_dscnt 0x0" ::: "memory");
        __builtin_amdgcn_wave_barrier();
    }
    // ---- normalize + store to [b, i, h*64+d] bf16 for final projection ----
#pragma unroll
    for (int r = 0; r < 8; r++) {
        const int i = i0 + r + half * 8;
        const float invl = 1.0f / l_run[r];
#pragma unroll
        for (int a = 0; a < 4; a++)
            attn_out[((size_t)b * SEQ + i) * DIM + h * DHEAD + a * 16 + c] =
                (__bf16)(acc[a][r] * invl);
    }
}

// ---------------------------------------------------------------------------
// Launch
// ---------------------------------------------------------------------------
extern "C" void kernel_launch(void* const* d_in, const int* in_sizes, int n_in,
                              void* d_out, int out_size, void* d_ws, size_t ws_size,
                              hipStream_t stream) {
    (void)in_sizes; (void)n_in; (void)out_size; (void)ws_size;
    const float* x       = (const float*)d_in[0];
    const float* gamma   = (const float*)d_in[1];
    const float* w_q     = (const float*)d_in[2];
    const float* w_kv    = (const float*)d_in[3];
    const float* null_kv = (const float*)d_in[4];
    const float* q_scale = (const float*)d_in[5];
    const float* k_scale = (const float*)d_in[6];
    const float* w_out   = (const float*)d_in[7];
    float* out = (float*)d_out;

    char* w = (char*)d_ws;
    size_t off = 0;
    auto alloc = [&](size_t bytes) { char* p = w + off; off = (off + bytes + 255) & ~(size_t)255; return p; };
    __bf16* xn      = (__bf16*)alloc((size_t)MTOT * DIM * 2);
    __bf16* wqT     = (__bf16*)alloc((size_t)DIM * DIM * 2);
    __bf16* wkvT    = (__bf16*)alloc((size_t)DIM * 2 * DIM * 2);
    __bf16* woutT   = (__bf16*)alloc((size_t)DIM * DIM * 2);
    __bf16* q_n     = (__bf16*)alloc((size_t)BATCH * HEADS * SEQ * DHEAD * 2);
    __bf16* k_n     = (__bf16*)alloc((size_t)BATCH * HEADS * JTOT * DHEAD * 2);
    __bf16* v_t     = (__bf16*)alloc((size_t)BATCH * HEADS * DHEAD * JPAD * 2 + 256);
    __bf16* attn_o  = (__bf16*)alloc((size_t)MTOT * DIM * 2);

    // weights -> bf16 transposed
    convT_kernel<<<(DIM * DIM + 255) / 256, 256, 0, stream>>>(w_q, wqT, DIM, DIM);
    convT_kernel<<<(DIM * 2 * DIM + 255) / 256, 256, 0, stream>>>(w_kv, wkvT, DIM, 2 * DIM);
    convT_kernel<<<(DIM * DIM + 255) / 256, 256, 0, stream>>>(w_out, woutT, DIM, DIM);
    // layernorm
    ln_kernel<<<MTOT, 256, 0, stream>>>(x, gamma, xn);
    // q projection (+ fused l2norm*q_scale): (8192/16)*(1024/64)=8192 waves
    gemm_wmma_kernel<1><<<8192 / 8, 256, 0, stream>>>(xn, wqT, DIM, DIM, nullptr,
                                                      q_n, nullptr, nullptr, q_scale, k_scale);
    // kv projection (+ fused k-norm / v-transpose): 16384 waves
    gemm_wmma_kernel<2><<<16384 / 8, 256, 0, stream>>>(xn, wkvT, DIM, 2 * DIM, nullptr,
                                                       nullptr, k_n, v_t, q_scale, k_scale);
    // null kv rows
    nullkv_kernel<<<(BATCH * HEADS * NNULL * 32 + 255) / 256, 256, 0, stream>>>(null_kv, k_scale, k_n, v_t);
    // attention: b*h*(2048/16) = 8192 waves
    attn_kernel<<<8192 / 8, 256, 0, stream>>>(q_n, k_n, v_t, attn_o);
    // output projection -> f32 d_out
    gemm_wmma_kernel<0><<<8192 / 8, 256, 0, stream>>>(attn_o, woutT, DIM, DIM, out,
                                                      nullptr, nullptr, nullptr, q_scale, k_scale);
}
